// GCN_68375879352699
// MI455X (gfx1250) — matrix operand order repaired and checked by reading, not compiled, not run on hardware
//
#include <hip/hip_runtime.h>
#include <hip/hip_bf16.h>
#include <stdint.h>

typedef __attribute__((ext_vector_type(16))) __bf16 v16bf;
typedef __attribute__((ext_vector_type(8)))  float  v8f;

#define B_  8
#define N_  1024
#define D_  512
#define BN_ (B_ * N_)
#define H_  2048   // 4*D

// ---------------- helpers ----------------

__device__ __forceinline__ unsigned short f2bf(float f) {
  union { float f; unsigned u; } v; v.f = f;
  unsigned u = v.u;
  return (unsigned short)((u + 0x7FFFu + ((u >> 16) & 1u)) >> 16); // RNE
}

union FragU { uint4 q[2]; v16bf v; };

// A-fragment 16x32 bf16 from LDS, row-major with row stride 32 elements.
// Layout: lane M = lane&15; elems 0-15 -> K = kb..kb+15 (matches frag_b),
// where kb = 16*(lane>=16).  Both halves are contiguous 16B -> ds_load_b128 pairs.
__device__ __forceinline__ v16bf frag_a(const unsigned short* p, int row, int lane) {
  int kb = (lane >> 4) * 16;
  FragU f;
  f.q[0] = *(const uint4*)(p + row * 32 + kb);
  f.q[1] = *(const uint4*)(p + row * 32 + kb + 8);
  return f.v;
}

// B-fragment 32x16 bf16 from a TRANSPOSED LDS store pT[n][32] (row stride 32).
// Contiguous k order (same as frag_a): lane N = lane&15; lanes 0-15 hold K=0..15, lanes 16-31 hold K=16..31.
__device__ __forceinline__ v16bf frag_b(const unsigned short* pT, int n, int lane) {
  int kb = (lane >> 4) * 16;
  const uint4* q = (const uint4*)(pT + n * 32 + kb);
  FragU f;
  f.q[0] = q[0];
  f.q[1] = q[1];
  return f.v;
}

__device__ __forceinline__ v8f wmma_bf16(v16bf a, v16bf b, v8f c) {
  // (neg_a, A, neg_b, B, c_mod, C, reuse_a, reuse_b)
  return __builtin_amdgcn_wmma_f32_16x16x32_bf16(false, a, false, b, (short)0, c, false, false);
}

__device__ __forceinline__ float wred_sum(float v) {
  #pragma unroll
  for (int m = 16; m >= 1; m >>= 1) v += __shfl_xor(v, m, 32);
  return v;
}
__device__ __forceinline__ float wred_max(float v) {
  #pragma unroll
  for (int m = 16; m >= 1; m >>= 1) v = fmaxf(v, __shfl_xor(v, m, 32));
  return v;
}

// ---------------- elementwise fp32 -> bf16 ----------------

__global__ void k_f2bf(const float* __restrict__ in, unsigned short* __restrict__ out, long n) {
  long i = (long)blockIdx.x * blockDim.x + threadIdx.x;
  long stride = (long)gridDim.x * blockDim.x;
  for (; i < n; i += stride) out[i] = f2bf(in[i]);
}

// ---------------- attention score projections: si = x.a1, sj = x.a2 ----------------

__global__ __launch_bounds__(256) void k_scores(
    const float* __restrict__ x, const float* __restrict__ w,
    float* __restrict__ si, float* __restrict__ sj) {
  int lane = threadIdx.x & 31;
  int row  = blockIdx.x * 8 + (threadIdx.x >> 5);
  const float4* xr = (const float4*)(x + (long)row * D_);
  const float4* w1 = (const float4*)(w);
  const float4* w2 = (const float4*)(w + D_);
  float a = 0.f, b = 0.f;
  #pragma unroll
  for (int k = 0; k < 4; ++k) {
    int idx = lane + k * 32;
    float4 xv = xr[idx], wa = w1[idx], wb = w2[idx];
    a += xv.x * wa.x + xv.y * wa.y + xv.z * wa.z + xv.w * wa.w;
    b += xv.x * wb.x + xv.y * wb.y + xv.z * wb.z + xv.w * wb.w;
  }
  a = wred_sum(a); b = wred_sum(b);
  if (lane == 0) { si[row] = a; sj[row] = b; }
}

// ---------------- layernorm (wave per row), dual fp32 + bf16 outputs ----------------

__global__ __launch_bounds__(256) void k_ln(
    const float* __restrict__ xin, const float* __restrict__ g, const float* __restrict__ bet,
    float* __restrict__ yf, unsigned short* __restrict__ yb) {
  int lane = threadIdx.x & 31;
  int row  = blockIdx.x * 8 + (threadIdx.x >> 5);
  const float4* xr = (const float4*)(xin + (long)row * D_);
  float4 v[4];
  float s = 0.f, s2 = 0.f;
  #pragma unroll
  for (int k = 0; k < 4; ++k) {
    v[k] = xr[lane + k * 32];
    s  += v[k].x + v[k].y + v[k].z + v[k].w;
    s2 += v[k].x * v[k].x + v[k].y * v[k].y + v[k].z * v[k].z + v[k].w * v[k].w;
  }
  s = wred_sum(s); s2 = wred_sum(s2);
  float mean = s * (1.0f / D_);
  float var  = s2 * (1.0f / D_) - mean * mean;
  float rs   = rsqrtf(var + 1e-5f);
  #pragma unroll
  for (int k = 0; k < 4; ++k) {
    int idx = lane + k * 32;
    float4 gv = ((const float4*)g)[idx];
    float4 bv = ((const float4*)bet)[idx];
    float4 o;
    o.x = (v[k].x - mean) * rs * gv.x + bv.x;
    o.y = (v[k].y - mean) * rs * gv.y + bv.y;
    o.z = (v[k].z - mean) * rs * gv.z + bv.z;
    o.w = (v[k].w - mean) * rs * gv.w + bv.w;
    ((float4*)(yf + (long)row * D_))[idx] = o;
    union { unsigned short h[4]; uint2 q; } pk;
    pk.h[0] = f2bf(o.x); pk.h[1] = f2bf(o.y); pk.h[2] = f2bf(o.z); pk.h[3] = f2bf(o.w);
    ((uint2*)(yb + (long)row * D_))[idx] = pk.q;
  }
}

// ---------------- fused masked-softmax attention + aggregation (flash-style) ----------------
// block = (16 target rows, one batch); 8 waves, wave w owns output cols [w*64, w*64+64)

__global__ __launch_bounds__(256) void k_attn(
    const float* __restrict__ xin, const unsigned short* __restrict__ xb,
    const float* __restrict__ si_g, const float* __restrict__ sj_g,
    const float* __restrict__ ab_p, const int* __restrict__ adj,
    float* __restrict__ hout) {
  __shared__ __align__(16) unsigned short s_xT[D_ * 32]; // x tile transposed [d][j], 32KB
  __shared__ __align__(16) unsigned short s_e[16 * 32];  // exp-score A tile (bf16)
  __shared__ float s_scale[16];
  __shared__ float s_sum[16];

  int w = threadIdx.x >> 5, lane = threadIdx.x & 31;
  int b  = blockIdx.y;
  int i0 = blockIdx.x * 16;
  long bN = (long)b * N_;
  float ab = ab_p[0];

  v8f z = {0.f, 0.f, 0.f, 0.f, 0.f, 0.f, 0.f, 0.f};
  v8f acc[4];
  #pragma unroll
  for (int t = 0; t < 4; ++t) acc[t] = z;

  float mreg[2] = {-1e30f, -1e30f};  // running max per owned row
  float sreg[2] = {0.f, 0.f};        // running sum per owned row
  float sii[2];
  sii[0] = si_g[bN + i0 + w];
  sii[1] = si_g[bN + i0 + 8 + w];

  for (int jt = 0; jt < N_; jt += 32) {
    __syncthreads();
    // stage x^T tile: 32 source rows j, all 512 dims, bf16
    for (int c = threadIdx.x; c < 2048; c += 256) {
      int j = c >> 6, dblk = c & 63;
      uint4 q = *(const uint4*)(xb + (bN + jt + j) * (long)D_ + dblk * 8);
      const unsigned short* hs = (const unsigned short*)&q;
      int dbase = dblk * 8;
      #pragma unroll
      for (int e = 0; e < 8; ++e) s_xT[(dbase + e) * 32 + j] = hs[e];
    }
    // online-softmax stats: wave w handles rows w and w+8, lane = j within tile
    #pragma unroll
    for (int h = 0; h < 2; ++h) {
      int row = h * 8 + w;
      int i   = i0 + row;
      int j   = jt + lane;
      int av  = adj[(bN + i) * (long)N_ + j];
      bool valid = (av != 0) || (i == j);
      float s = sii[h] + sj_g[bN + j] + ab;
      s = (s >= 0.f) ? s : 0.2f * s;   // LeakyReLU(0.2)
      s = valid ? s : -1e30f;
      float m    = wred_max(s);
      float mnew = fmaxf(mreg[h], m);
      float scl  = __expf(mreg[h] - mnew);
      float e    = valid ? __expf(s - mnew) : 0.f;
      s_e[row * 32 + lane] = f2bf(e);  // e in [0,1] -> bf16 safe
      float se = wred_sum(e);
      sreg[h] = sreg[h] * scl + se;
      mreg[h] = mnew;
      if (lane == 0) s_scale[row] = scl;
    }
    __syncthreads();
    // rescale accumulators then accumulate this tile: acc += E(16x32) @ X(32x64)
    int hi = lane >> 4;
    float sc[8];
    #pragma unroll
    for (int r = 0; r < 8; ++r) sc[r] = s_scale[r + 8 * hi];
    #pragma unroll
    for (int t = 0; t < 4; ++t)
      #pragma unroll
      for (int r = 0; r < 8; ++r) acc[t][r] *= sc[r];
    v16bf afr = frag_a(s_e, lane & 15, lane);
    #pragma unroll
    for (int t = 0; t < 4; ++t) {
      v16bf bfr = frag_b(s_xT, w * 64 + t * 16 + (lane & 15), lane);
      acc[t] = wmma_bf16(afr, bfr, acc[t]);
    }
  }
  if (lane == 0) { s_sum[w] = sreg[0]; s_sum[8 + w] = sreg[1]; }
  __syncthreads();
  // epilogue: normalize by softmax sum, add residual x
  int hi = lane >> 4;
  #pragma unroll
  for (int t = 0; t < 4; ++t) {
    #pragma unroll
    for (int r = 0; r < 8; ++r) {
      int rl = r + 8 * hi;
      long gi = (bN + i0 + rl) * (long)D_ + w * 64 + t * 16 + (lane & 15);
      hout[gi] = acc[t][r] / s_sum[rl] + xin[gi];
    }
  }
}

// ---------------- bf16 GEMM: C = A[MxK] @ B[KxN]; fused GELU->bf16 or bias+residual->fp32
// Block tile 256x64, 8 waves, wave = 32Mx64N (2 A-frags x 4 B-frags = 8 WMMA / K-step).
// Register double-buffering: next K-tile's global loads are issued before the WMMAs.

__global__ __launch_bounds__(256) void k_gemm(
    const unsigned short* __restrict__ A, const unsigned short* __restrict__ Bm,
    int M, int N, int K,
    const float* __restrict__ bias, const float* __restrict__ res,
    float* __restrict__ outF, unsigned short* __restrict__ outB, int mode) {
  __shared__ __align__(16) unsigned short sA[256 * 32];  // A tile row-major, 16KB
  __shared__ __align__(16) unsigned short sBT[64 * 32];  // B tile transposed, 4KB
  int w = threadIdx.x >> 5, lane = threadIdx.x & 31;
  long rb = (long)blockIdx.x * 256, cb = (long)blockIdx.y * 64;

  v8f z = {0.f, 0.f, 0.f, 0.f, 0.f, 0.f, 0.f, 0.f};
  v8f acc[2][4];
  #pragma unroll
  for (int u = 0; u < 2; ++u)
    #pragma unroll
    for (int t = 0; t < 4; ++t) acc[u][t] = z;

  // staging maps
  int rowA = threadIdx.x >> 2, chA = (threadIdx.x & 3) * 8; // 4 threads/row, 16B each
  int kB = threadIdx.x >> 3, nb = threadIdx.x & 7;          // 8 threads/k-row

  // preload first K-tile into registers
  uint4 ra[4];
  uint4 rbq;
  #pragma unroll
  for (int rr = 0; rr < 4; ++rr)
    ra[rr] = *(const uint4*)(A + (rb + rr * 64 + rowA) * (long)K + chA);
  rbq = *(const uint4*)(Bm + (long)kB * N + cb + nb * 8);

  for (int k0 = 0; k0 < K; k0 += 32) {
    __syncthreads();                    // previous tile's WMMAs done
    // registers -> LDS
    #pragma unroll
    for (int rr = 0; rr < 4; ++rr)
      *(uint4*)(sA + (rr * 64 + rowA) * 32 + chA) = ra[rr];
    {
      const unsigned short* hs = (const unsigned short*)&rbq;
      #pragma unroll
      for (int e = 0; e < 8; ++e) sBT[(nb * 8 + e) * 32 + kB] = hs[e];
    }
    __syncthreads();
    // issue next tile's global loads; they overlap the WMMAs below
    int kn = k0 + 32;
    if (kn < K) {
      #pragma unroll
      for (int rr = 0; rr < 4; ++rr)
        ra[rr] = *(const uint4*)(A + (rb + rr * 64 + rowA) * (long)K + kn + chA);
      rbq = *(const uint4*)(Bm + (long)(kn + kB) * N + cb + nb * 8);
      if (kn + 32 < K)                  // far-ahead hint: global_prefetch_b8
        __builtin_prefetch(A + (rb + rowA) * (long)K + kn + 32, 0, 1);
    }
    // 8 WMMAs: 2 A-frags x 4 B-frags
    v16bf af0 = frag_a(sA, w * 32 + (lane & 15), lane);
    v16bf af1 = frag_a(sA, w * 32 + 16 + (lane & 15), lane);
    #pragma unroll
    for (int t = 0; t < 4; ++t) {
      v16bf bfr = frag_b(sBT, t * 16 + (lane & 15), lane);
      acc[0][t] = wmma_bf16(af0, bfr, acc[0][t]);
      acc[1][t] = wmma_bf16(af1, bfr, acc[1][t]);
    }
  }
  int hi = lane >> 4;
  #pragma unroll
  for (int u = 0; u < 2; ++u) {
    #pragma unroll
    for (int t = 0; t < 4; ++t) {
      #pragma unroll
      for (int r = 0; r < 8; ++r) {
        long i = rb + w * 32 + u * 16 + r + 8 * hi;
        long n = cb + t * 16 + (lane & 15);
        float v = acc[u][t][r] + bias[n];
        if (mode == 0) { // exact GELU, emit bf16 activation
          v = 0.5f * v * (1.0f + erff(v * 0.70710678118f));
          outB[i * N + n] = f2bf(v);
        } else {         // bias + residual, fp32 (pre-LN)
          outF[i * N + n] = v + res[i * N + n];
        }
      }
    }
  }
}

// ---------------- host-side orchestration ----------------

extern "C" void kernel_launch(void* const* d_in, const int* in_sizes, int n_in,
                              void* d_out, int out_size, void* d_ws, size_t ws_size,
                              hipStream_t stream) {
  (void)in_sizes; (void)n_in; (void)out_size; (void)ws_size;
  const float* x0  = (const float*)d_in[0];
  const int*   adj = (const int*)d_in[1];
  const float* aw  = (const float*)d_in[2];
  const float* abv = (const float*)d_in[3];
  const float* W1  = (const float*)d_in[4];
  const float* b1  = (const float*)d_in[5];
  const float* W2  = (const float*)d_in[6];
  const float* b2  = (const float*)d_in[7];
  const float* g1  = (const float*)d_in[8];
  const float* be1 = (const float*)d_in[9];
  const float* g2  = (const float*)d_in[10];
  const float* be2 = (const float*)d_in[11];

  char* ws = (char*)d_ws;
  size_t off = 0;
  auto alloc = [&](size_t bytes) -> char* {
    char* p = ws + off; off += (bytes + 255) & ~(size_t)255; return p;
  };
  unsigned short* xb  = (unsigned short*)alloc((size_t)BN_ * D_ * 2);
  unsigned short* hb  = (unsigned short*)alloc((size_t)BN_ * D_ * 2);
  float* hinf = (float*)alloc((size_t)BN_ * D_ * 4);
  float* hf   = (float*)alloc((size_t)BN_ * D_ * 4);
  float* xf   = (float*)alloc((size_t)BN_ * D_ * 4);
  float* si   = (float*)alloc((size_t)BN_ * 4);
  float* sj   = (float*)alloc((size_t)BN_ * 4);
  unsigned short* C1  = (unsigned short*)alloc((size_t)BN_ * H_ * 2);
  unsigned short* W1b = (unsigned short*)alloc((size_t)3 * D_ * H_ * 2);
  unsigned short* W2b = (unsigned short*)alloc((size_t)3 * H_ * D_ * 2);

  k_f2bf<<<512, 256, 0, stream>>>(W1, W1b, (long)3 * D_ * H_);
  k_f2bf<<<512, 256, 0, stream>>>(W2, W2b, (long)3 * H_ * D_);
  k_f2bf<<<512, 256, 0, stream>>>(x0, xb, (long)BN_ * D_);

  const float* xin = x0;
  for (int l = 0; l < 3; ++l) {
    k_scores<<<BN_ / 8, 256, 0, stream>>>(xin, aw + (size_t)l * 2 * D_, si, sj);
    k_attn<<<dim3(N_ / 16, B_), 256, 0, stream>>>(xin, xb, si, sj, abv + l, adj, hinf);
    k_ln<<<BN_ / 8, 256, 0, stream>>>(hinf, g1 + l * D_, be1 + l * D_, hf, hb);
    k_gemm<<<dim3(BN_ / 256, H_ / 64), 256, 0, stream>>>(
        hb, W1b + (size_t)l * D_ * H_, BN_, H_, D_, b1 + (size_t)l * H_,
        nullptr, nullptr, C1, 0);
    k_gemm<<<dim3(BN_ / 256, D_ / 64), 256, 0, stream>>>(
        C1, W2b + (size_t)l * H_ * D_, BN_, D_, H_, b2 + (size_t)l * D_,
        hf, hinf, nullptr, 1);
    float* xo = (l == 2) ? (float*)d_out : xf;
    k_ln<<<BN_ / 8, 256, 0, stream>>>(hinf, g2 + l * D_, be2 + l * D_, xo, xb);
    xin = xo;
  }
}
